// PointCFPFusionModule_12807592477405
// MI455X (gfx1250) — matrix operand order repaired and checked by reading, not compiled, not run on hardware
//
#include <hip/hip_runtime.h>
#include <math.h>
#include <stdint.h>

#define N_HI 65536
#define N_LO 4096
#define CCH  128
#define MIDC 32
#define NB   4

typedef __attribute__((ext_vector_type(2)))  float  v2f;
typedef __attribute__((ext_vector_type(8)))  float  v8f;
typedef __attribute__((ext_vector_type(8)))  __bf16 v8bf;
typedef __attribute__((ext_vector_type(16))) __bf16 v16bf;
typedef __attribute__((ext_vector_type(4)))  unsigned int v4u;
typedef __attribute__((ext_vector_type(8)))  int v8i;
typedef __attribute__((ext_vector_type(4)))  int v4i;

static __device__ __forceinline__ unsigned short f2bf_bits(float f) {
  __bf16 h = (__bf16)f;
  return __builtin_bit_cast(unsigned short, h);
}

static __device__ __forceinline__ int batch_of(int i, int o0, int o1, int o2, int o3) {
  int b = (o0 <= i) + (o1 <= i) + (o2 <= i) + (o3 <= i);
  return b > 3 ? 3 : b;
}

// ---------------------------------------------------------------------------
// Zero accumulators in workspace (re-run every launch for determinism).
// ---------------------------------------------------------------------------
__global__ void k_zero(float* gf, float* bnSum, float* bnSqs) {
  int t = threadIdx.x;            // 256 threads, 1 block
  gf[t] = 0.f;
  gf[t + 256] = 0.f;
  if (t < 128) { bnSum[t] = 0.f; bnSqs[t] = 0.f; }
}

// ---------------------------------------------------------------------------
// Pack Ws1[0:128][:] and Wf[0:256][:] into WMMA B-matrix per-lane layout.
// packed[kt][nt][lane][e] = W[(kt*32 + 8*(lane>=16) + (e&7) + 16*(e>>3))*128 + nt*16 + (lane&15)]
// ---------------------------------------------------------------------------
__global__ void k_pack(const float* __restrict__ Ws1, const float* __restrict__ Wf,
                       unsigned short* __restrict__ pWs1, unsigned short* __restrict__ pWf) {
  int g = blockIdx.x * 256 + threadIdx.x;   // 192 blocks -> 49152 threads
  if (g < 16384) {                          // Ws1 top 128 rows: kt in 0..3
    int e    = g & 15;
    int lane = (g >> 4) & 31;
    int nt   = (g >> 9) & 7;
    int kt   = g >> 12;
    int K = kt * 32 + ((lane >> 4) & 1) * 8 + (e & 7) + (e >> 3) * 16;
    int N = nt * 16 + (lane & 15);
    pWs1[g] = f2bf_bits(Ws1[K * CCH + N]);
  } else if (g < 49152) {                   // Wf 256 rows: kt in 0..7
    int q    = g - 16384;
    int e    = q & 15;
    int lane = (q >> 4) & 31;
    int nt   = (q >> 9) & 7;
    int kt   = q >> 12;
    int K = kt * 32 + ((lane >> 4) & 1) * 8 + (e & 7) + (e >> 3) * 16;
    int N = nt * 16 + (lane & 15);
    pWf[q] = f2bf_bits(Wf[K * CCH + N]);
  }
}

// ---------------------------------------------------------------------------
// Nearest neighbor via V_WMMA_F32_16X16X4_F32.
// low_res_coord (48KB) is staged Global->LDS with one Tensor Data Mover
// descriptor (D#) issued by wave 0, tracked by TENSORcnt. Derived arrays
// (-2*x,-2*y,-2*z, ||.||^2) are then built LDS->LDS.
// Block: 256 thr = 8 waves; wave handles 16 high points; 512 blocks.
// ---------------------------------------------------------------------------
__global__ void k_nn(const float* __restrict__ hc, const float* __restrict__ lc,
                     int* __restrict__ idxOut) {
  extern __shared__ char smemA[];
  float* rawc = (float*)(smemA);            // [4096*3] f32  (48KB) via TDM
  float* lax  = (float*)(smemA + 49152);
  float* lay  = (float*)(smemA + 65536);
  float* laz  = (float*)(smemA + 81920);
  float* l2   = (float*)(smemA + 98304);    // total 96KB + this 16KB = 114688B

  int t = threadIdx.x;

  // --- TDM: DMA 12288 f32 (one row) from lc into LDS rawc ---
  if (t < 32) {
    unsigned long long ga = (unsigned long long)(uintptr_t)lc;
    unsigned int lds_off = (unsigned int)(uintptr_t)rawc;  // flat[31:0] == LDS addr
    v4u g0;
    g0[0] = 1u;                                        // count=1, user descriptor
    g0[1] = lds_off;                                   // lds_addr
    g0[2] = (unsigned int)(ga & 0xffffffffull);        // global_addr[31:0]
    g0[3] = (unsigned int)((ga >> 32) & 0x01fffffful)  // global_addr[56:32]
          | (2u << 30);                                // type=2 ("image")
    v8i g1;
    g1[0] = (int)(2u << 16);        // workgroup_mask=0, data_size=2 (4B)
    g1[1] = (int)(12288u << 16);    // tensor_dim0[15:0]  (=12288)
    g1[2] = (int)(1u << 16);        // tensor_dim0[31:16]=0, tensor_dim1[15:0]=1
    g1[3] = (int)(12288u << 16);    // tensor_dim1[31:16]=0, tile_dim0=12288
    g1[4] = 1;                      // tile_dim1=1, tile_dim2=0
    g1[5] = 12288;                  // tensor_dim0_stride[31:0]
    g1[6] = (int)(12288u << 16);    // stride0[47:32]=0, tensor_dim1_stride[15:0]
    g1[7] = 0;                      // tensor_dim1_stride[47:16]=0
    v4i gz4 = { 0, 0, 0, 0 };
    v8i gz8 = { 0, 0, 0, 0, 0, 0, 0, 0 };
    __builtin_amdgcn_tensor_load_to_lds(g0, g1, gz4, gz4, gz8, 0);
    __builtin_amdgcn_s_wait_tensorcnt(0);
  }
  __syncthreads();

  for (int j = t; j < N_LO; j += 256) {
    float x = rawc[j * 3 + 0], y = rawc[j * 3 + 1], z = rawc[j * 3 + 2];
    lax[j] = -2.f * x; lay[j] = -2.f * y; laz[j] = -2.f * z;
    l2[j] = x * x + y * y + z * z;
  }
  __syncthreads();

  int w = t >> 5, lane = t & 31;
  int m = lane & 15, hb = lane >> 4;          // hb: half-wave selector
  int n = blockIdx.x * 128 + w * 16 + m;      // this lane's high point (col N)

  float hx = hc[n * 3 + 0], hy = hc[n * 3 + 1], hz = hc[n * 3 + 2];
  v2f bv;
  bv.x = hb ? hz : hx;   // B: VGPR0 -> K=0 (lanes 0-15) / K=2 (lanes 16-31)
  bv.y = hb ? 0.f : hy;  //    VGPR1 -> K=1 / K=3(pad)

  float minv = 3.0e38f;
  int   mini = 0;
  for (int tile = 0; tile < N_LO / 16; ++tile) {
    int mm = tile * 16 + m;
    v2f av;
    av.x = hb ? laz[mm] : lax[mm];   // A: VGPR0 -> K=0 / K=2 ; VGPR1 -> K=1 / K=3
    av.y = hb ? 0.f : lay[mm];
    v8f c = 0.f;
    v8f d = __builtin_amdgcn_wmma_f32_16x16x4_f32(false, av, false, bv,
                                                  (short)0, c, false, false);
    const float* l2p = l2 + tile * 16 + hb * 8;
#pragma unroll
    for (int v = 0; v < 8; ++v) {
      float sc = d[v] + l2p[v];
      int li = tile * 16 + hb * 8 + v;
      bool lt = sc < minv;
      minv = lt ? sc : minv;
      mini = lt ? li : mini;
    }
  }
  float ov = __shfl_xor(minv, 16, 32);
  int   oi = __shfl_xor(mini, 16, 32);
  bool take = (ov < minv) || (ov == minv && oi < mini);
  minv = take ? ov : minv;
  mini = take ? oi : mini;
  if (hb == 0) idxOut[n] = mini;
}

// ---------------------------------------------------------------------------
// Segment sum of low_res_feat[idx[i]] into gf[4][128] (register-accumulated,
// flushed via global atomics on batch change). 64 blocks x 256 thr.
// ---------------------------------------------------------------------------
__global__ void k_segsum(const float* __restrict__ lf, const int* __restrict__ idx,
                         const int* __restrict__ offs, float* __restrict__ gf) {
  int t = threadIdx.x;
  int ch = t & 127, sub = t >> 7;
  int base = blockIdx.x * 1024;
  int o0 = offs[0], o1 = offs[1], o2 = offs[2], o3 = offs[3];
  int curb = -1;
  float acc = 0.f;
  for (int p = sub; p < 1024; p += 2) {
    int i = base + p;
    int b = batch_of(i, o0, o1, o2, o3);
    if (b != curb) {
      if (curb >= 0) atomicAdd(&gf[curb * CCH + ch], acc);
      curb = b; acc = 0.f;
    }
    acc += lf[(size_t)idx[i] * CCH + ch];
  }
  if (curb >= 0) atomicAdd(&gf[curb * CCH + ch], acc);
}

// ---------------------------------------------------------------------------
// Channel-attention MLP on [4,128]: sigmoid(relu(g@W1+b1)@W2+b2). 1 block/128 thr.
// ---------------------------------------------------------------------------
__global__ void k_chan(const float* __restrict__ gf, const int* __restrict__ offs,
                       const float* __restrict__ W1, const float* __restrict__ b1,
                       const float* __restrict__ W2, const float* __restrict__ b2,
                       float* __restrict__ cwOut) {
  __shared__ float gm[NB * CCH];
  __shared__ float hS[NB * MIDC];
  int t = threadIdx.x;  // 128
  int o0 = offs[0], o1 = offs[1], o2 = offs[2], o3 = offs[3];
  float cnt[4] = { (float)max(o0, 1), (float)max(o1 - o0, 1),
                   (float)max(o2 - o1, 1), (float)max(o3 - o2, 1) };
  for (int b = 0; b < NB; ++b) gm[b * CCH + t] = gf[b * CCH + t] / cnt[b];
  __syncthreads();
  {
    int bb = t >> 5, j = t & 31;
    float s = b1[j];
    for (int c = 0; c < CCH; ++c) s += gm[bb * CCH + c] * W1[c * MIDC + j];
    hS[bb * MIDC + j] = fmaxf(s, 0.f);
  }
  __syncthreads();
  for (int bb = 0; bb < NB; ++bb) {
    float s = b2[t];
    for (int j = 0; j < MIDC; ++j) s += hS[bb * MIDC + j] * W2[j * CCH + t];
    cwOut[bb * CCH + t] = 1.f / (1.f + __expf(-s));
  }
}

// ---------------------------------------------------------------------------
// Main fused kernel: channel refine -> spatial attention (WMMA bf16 K=128 +
// rank-3 coord fold) -> fusion GEMM (WMMA bf16 K=256) -> fused to d_out + BN
// partial sums. Block: 256 thr = 8 waves; wave = 16 points; 512 blocks.
// ---------------------------------------------------------------------------
__global__ void k_main(const float* __restrict__ hf, const float* __restrict__ hc,
                       const int* __restrict__ offs, const float* __restrict__ cw,
                       const float* __restrict__ Ws1, const float* __restrict__ bs1,
                       const float* __restrict__ Ws2, const float* __restrict__ bs2,
                       const float* __restrict__ bff,
                       const unsigned short* __restrict__ pWs1u,
                       const unsigned short* __restrict__ pWfu,
                       float* __restrict__ outFused,
                       float* __restrict__ bnSum, float* __restrict__ bnSqs) {
  extern __shared__ char smem[];
  __bf16* crS = (__bf16*)(smem);            // [8][16][128] bf16  (32KB)
  __bf16* a2S = (__bf16*)(smem + 32768);    // [8][16][256] bf16  (64KB)
  float*  h1S = (float*)(smem + 98304);     // [8][16][128] f32   (64KB)
  float*  cS  = (float*)(smem + 163840);    // [8][48]            (1.5KB)
  float*  bnS = (float*)(smem + 165376);    // [256]              (1KB)
  const __bf16* pWs1 = (const __bf16*)pWs1u;
  const __bf16* pWf  = (const __bf16*)pWfu;

  int t = threadIdx.x;

  // Pull the shared packed-weight region (96KB, contiguous) toward this WGP.
  __builtin_prefetch(pWs1u + (size_t)t * 64, 0, 3);            // 32KB region
  __builtin_prefetch(pWfu  + (size_t)t * 64, 0, 3);            // 64KB region
  __builtin_prefetch(pWfu  + 16384 + (size_t)t * 64, 0, 3);

  int w = t >> 5, lane = t & 31;
  int m = lane & 15, hb = lane >> 4;
  int base = blockIdx.x * 128 + w * 16;
  int i0 = base + m;

  bnS[t] = 0.f;

  int o0 = offs[0], o1 = offs[1], o2 = offs[2], o3 = offs[3];
  int b = batch_of(i0, o0, o1, o2, o3);

  // Stage channel-refined features (bf16) + coords.
  const float* hfp = hf + (size_t)i0 * CCH;
  const float* cwp = cw + b * CCH;
  for (int c = hb * 64; c < hb * 64 + 64; ++c)
    crS[(w * 16 + m) * CCH + c] = (__bf16)(hfp[c] * cwp[c]);
  if (hb == 0) {
    cS[w * 48 + m]      = hc[i0 * 3 + 0];
    cS[w * 48 + 16 + m] = hc[i0 * 3 + 1];
    cS[w * 48 + 32 + m] = hc[i0 * 3 + 2];
  }
  __syncthreads();

  // GEMM1: h1 = relu( [cr | coord] @ Ws1 + bs1 ),  K=128 WMMA + rank-3 fold.
  for (int nt = 0; nt < 8; ++nt) {
    int n = nt * 16 + m;
    float bias = bs1[n];
    float wr0 = Ws1[128 * CCH + n], wr1 = Ws1[129 * CCH + n], wr2 = Ws1[130 * CCH + n];
    v8f acc;
#pragma unroll
    for (int v = 0; v < 8; ++v) {
      int M = v + hb * 8;
      acc[v] = bias + cS[w * 48 + M] * wr0 + cS[w * 48 + 16 + M] * wr1
                    + cS[w * 48 + 32 + M] * wr2;
    }
#pragma unroll
    for (int kt = 0; kt < 4; ++kt) {
      const __bf16* arow = crS + (w * 16 + m) * CCH + kt * 32 + hb * 8;
      v8bf lo = *(const v8bf*)(arow);
      v8bf hi = *(const v8bf*)(arow + 16);
      v16bf A;
#pragma unroll
      for (int j = 0; j < 8; ++j) { A[j] = lo[j]; A[8 + j] = hi[j]; }
      v16bf B = *(const v16bf*)(pWs1 + ((kt * 8 + nt) * 32 + lane) * 16);
      acc = __builtin_amdgcn_wmma_f32_16x16x32_bf16(false, A, false, B,
                                                    (short)0, acc, false, false);
    }
#pragma unroll
    for (int v = 0; v < 8; ++v) {
      int M = v + hb * 8;
      h1S[(w * 16 + M) * CCH + n] = fmaxf(acc[v], 0.f);
    }
  }
  __syncthreads();

  // Spatial weight: sw = sigmoid(h1 . Ws2 + bs2), split over lane halves.
  float sw;
  {
    float s = 0.f;
    const float* hrow = h1S + (w * 16 + m) * CCH + hb * 64;
    for (int c = 0; c < 64; ++c) s += hrow[c] * Ws2[hb * 64 + c];
    s += __shfl_xor(s, 16, 32);
    sw = 1.f / (1.f + __expf(-(s + bs2[0])));
  }

  // Stage A2 = [high_feat | cr*sw] as bf16.
  for (int c = hb * 64; c < hb * 64 + 64; ++c) {
    a2S[(w * 16 + m) * 256 + c] = (__bf16)hfp[c];
    float cr = (float)crS[(w * 16 + m) * CCH + c];
    a2S[(w * 16 + m) * 256 + 128 + c] = (__bf16)(cr * sw);
  }
  __syncthreads();

  // GEMM2: fused = A2 @ Wf + bf,  K=256 WMMA; write fused + BN partials.
  for (int nt = 0; nt < 8; ++nt) {
    int n = nt * 16 + m;
    float bias = bff[n];
    v8f acc;
#pragma unroll
    for (int v = 0; v < 8; ++v) acc[v] = bias;
#pragma unroll
    for (int kt = 0; kt < 8; ++kt) {
      const __bf16* arow = a2S + (w * 16 + m) * 256 + kt * 32 + hb * 8;
      v8bf lo = *(const v8bf*)(arow);
      v8bf hi = *(const v8bf*)(arow + 16);
      v16bf A;
#pragma unroll
      for (int j = 0; j < 8; ++j) { A[j] = lo[j]; A[8 + j] = hi[j]; }
      v16bf B = *(const v16bf*)(pWf + ((kt * 8 + nt) * 32 + lane) * 16);
      acc = __builtin_amdgcn_wmma_f32_16x16x32_bf16(false, A, false, B,
                                                    (short)0, acc, false, false);
    }
    float s = 0.f, s2 = 0.f;
#pragma unroll
    for (int v = 0; v < 8; ++v) {
      float f = acc[v];
      int M = v + hb * 8;
      outFused[(size_t)(base + M) * CCH + n] = f;
      s += f; s2 += f * f;
    }
    s  += __shfl_xor(s, 16, 32);
    s2 += __shfl_xor(s2, 16, 32);
    if (hb == 0) { atomicAdd(&bnS[n], s); atomicAdd(&bnS[128 + n], s2); }
  }
  __syncthreads();
  if (t < 128) atomicAdd(&bnSum[t], bnS[t]);
  else         atomicAdd(&bnSqs[t - 128], bnS[t]);
}

// ---------------------------------------------------------------------------
// BN finalize: per-channel scale/shift, then elementwise normalize + relu.
// ---------------------------------------------------------------------------
__global__ void k_bnstats(const float* __restrict__ bnSum, const float* __restrict__ bnSqs,
                          const float* __restrict__ gamma, const float* __restrict__ beta,
                          float* __restrict__ ss) {
  int t = threadIdx.x;  // 128
  float inv_n = 1.f / (float)N_HI;
  float mean = bnSum[t] * inv_n;
  float var = bnSqs[t] * inv_n - mean * mean;
  float rs = rsqrtf(var + 1e-5f);
  float sc = gamma[t] * rs;
  ss[t] = sc;
  ss[128 + t] = beta[t] - mean * sc;
}

__global__ void k_bnapply(float* __restrict__ out, const float* __restrict__ ss) {
  size_t gid = (size_t)blockIdx.x * 256 + threadIdx.x;
  int ch = (int)(gid & 127);
  float f = out[gid];
  out[gid] = fmaxf(f * ss[ch] + ss[128 + ch], 0.f);
}

// ---------------------------------------------------------------------------
extern "C" void kernel_launch(void* const* d_in, const int* in_sizes, int n_in,
                              void* d_out, int out_size, void* d_ws, size_t ws_size,
                              hipStream_t stream) {
  (void)in_sizes; (void)n_in; (void)out_size; (void)ws_size;
  const float* hf   = (const float*)d_in[0];
  const float* hc   = (const float*)d_in[1];
  const float* lf   = (const float*)d_in[2];
  const float* lc   = (const float*)d_in[3];
  const int*   offs = (const int*)  d_in[4];
  const float* W1   = (const float*)d_in[5];
  const float* b1   = (const float*)d_in[6];
  const float* W2   = (const float*)d_in[7];
  const float* b2   = (const float*)d_in[8];
  const float* Ws1  = (const float*)d_in[9];
  const float* bs1  = (const float*)d_in[10];
  const float* Ws2  = (const float*)d_in[11];
  const float* bs2  = (const float*)d_in[12];
  const float* Wf   = (const float*)d_in[13];
  const float* bff  = (const float*)d_in[14];
  const float* gamma= (const float*)d_in[15];
  const float* beta = (const float*)d_in[16];
  float* out = (float*)d_out;

  char* ws = (char*)d_ws;
  int*   idx   = (int*)  (ws + 0);        // 65536 * 4  = 262144
  float* gf    = (float*)(ws + 262144);   // 512 * 4    -> 264192
  float* cwbuf = (float*)(ws + 264192);   // 512 * 4    -> 266240
  float* bnSum = (float*)(ws + 266240);   // 128 * 4    -> 266752
  float* bnSqs = (float*)(ws + 266752);   // 128 * 4    -> 267264
  float* ss    = (float*)(ws + 267264);   // 256 * 4    -> 268288
  unsigned short* pWs1 = (unsigned short*)(ws + 268288); // 16384*2 -> 301056
  unsigned short* pWf  = (unsigned short*)(ws + 301056); // 32768*2 -> 366592

  k_zero<<<1, 256, 0, stream>>>(gf, bnSum, bnSqs);
  k_pack<<<192, 256, 0, stream>>>(Ws1, Wf, pWs1, pWf);
  k_nn<<<N_HI / 128, 256, 114688, stream>>>(hc, lc, idx);
  k_segsum<<<N_HI / 1024, 256, 0, stream>>>(lf, idx, offs, gf);
  k_chan<<<1, 128, 0, stream>>>(gf, offs, W1, b1, W2, b2, cwbuf);
  k_main<<<N_HI / 128, 256, 166400, stream>>>(hf, hc, offs, cwbuf, Ws1, bs1, Ws2,
                                              bs2, bff, pWs1, pWf, out, bnSum, bnSqs);
  k_bnstats<<<1, 128, 0, stream>>>(bnSum, bnSqs, gamma, beta, ss);
  k_bnapply<<<(N_HI * CCH) / 256, 256, 0, stream>>>(out, ss);
}
